// Expert_35948876267749
// MI455X (gfx1250) — matrix operand order repaired and checked by reading, not compiled, not run on hardware
//
#include <hip/hip_runtime.h>

// MoE grouped FFN: out = relu(x @ fc1^T) @ fc2^T per expert.
// E=8, T=2048, D=1024, H=4096. fp32 in/out; compute = f16 WMMA, f32 accum.
//
// Pipeline:
//   cvt x/fc1/fc2 -> f16 in ws, then two TN GEMMs.
//   GEMM tiles fed with GLOBAL_LOAD_ASYNC_TO_LDS_B128 (ASYNCcnt) into
//   double-buffered LDS; software pipeline peeled so the steady-state loop
//   is branch-free; 8 v_wmma_f32_16x16x32_f16 per wave per K-step.
//
// Workspace layout (bytes):
//   xh  : E*T*D f16 =  33,554,432
//   w1h : E*H*D f16 =  67,108,864
//   w2h : E*D*H f16 =  67,108,864
//   h   : E*T*H f16 = 134,217,728   (fits in 192MB L2 between the GEMMs)

typedef __attribute__((ext_vector_type(16))) _Float16 v16h;
typedef __attribute__((ext_vector_type(8)))  _Float16 v8h;
typedef __attribute__((ext_vector_type(4)))  _Float16 v4h;
typedef __attribute__((ext_vector_type(8)))  float    v8f;
typedef __attribute__((ext_vector_type(4)))  float    v4f;

__global__ __launch_bounds__(256)
void cvt_f32_to_f16(const float* __restrict__ in, _Float16* __restrict__ out, size_t n4) {
    size_t i = (size_t)blockIdx.x * blockDim.x + threadIdx.x;
    if (i >= n4) return;
    v4f f = *(const v4f*)(in + i * 4);
    v4h h;
#pragma unroll
    for (int k = 0; k < 4; ++k) h[k] = (_Float16)f[k];
    *(v4h*)(out + i * 4) = h;
}

// Async copy of 16B per lane from global (SADDR base + 32-bit lane offset,
// GVS mode) directly into LDS at per-lane LDS byte address `lds_off`.
// Tracked by ASYNCcnt.
__device__ __forceinline__ void async_ld_b128(unsigned lds_off, unsigned goff,
                                              unsigned long long base) {
    asm volatile("global_load_async_to_lds_b128 %0, %1, %2"
                 :: "v"(lds_off), "v"(goff), "s"(base)
                 : "memory");
}
__device__ __forceinline__ void wait_async_le4() {
    asm volatile("s_wait_asynccnt 0x4" ::: "memory");
}
__device__ __forceinline__ void wait_async_0() {
    asm volatile("s_wait_asynccnt 0x0" ::: "memory");
}

// C[M,N] = act(A[M,K] * W[N,K]^T), row-major, expert = blockIdx.z.
// Block tile 128x128, BK=32 (one WMMA K step). 8 waves: 4(M) x 2(N);
// each wave owns 32x64 = 2x4 WMMA tiles. Requires K/32 even and >= 4.
template <bool RELU, typename OutT>
__global__ __launch_bounds__(256)
void gemm_tn(const _Float16* __restrict__ Ag, const _Float16* __restrict__ Wg,
             OutT* __restrict__ Cg, int M, int N, int K) {
    constexpr int BM = 128, BN = 128, BK = 32;
    constexpr int LDT = 40;  // padded LDS row stride in halves (80B): conflict-free

    __shared__ _Float16 lA[2][BM * LDT];
    __shared__ _Float16 lW[2][BN * LDT];

    const int tid  = threadIdx.x;
    const int lane = tid & 31;
    const int wave = tid >> 5;
    const int wm   = wave & 3;   // M offset wm*32
    const int wn   = wave >> 2;  // N offset wn*64

    const size_t e = blockIdx.z;
    const _Float16* A = Ag + e * (size_t)M * K;
    const _Float16* W = Wg + e * (size_t)N * K;
    OutT*           C = Cg + e * (size_t)M * N;

    const int m0 = blockIdx.y * BM;
    const int n0 = blockIdx.x * BN;

    // ---- staging coordinates: 512 16B-chunks per 128x32 tile, 2 per thread
    int rowc[2], kcc[2];
#pragma unroll
    for (int j = 0; j < 2; ++j) {
        int c   = tid + j * 256;
        rowc[j] = c >> 2;
        kcc[j]  = (c & 3) * 8;
    }
    // Global byte offsets (relative to per-expert base) at k0 = 0.
    unsigned aoff[2], woff[2];
#pragma unroll
    for (int j = 0; j < 2; ++j) {
        aoff[j] = (unsigned)((((size_t)(m0 + rowc[j]) * K) + kcc[j]) * 2);
        woff[j] = (unsigned)((((size_t)(n0 + rowc[j]) * K) + kcc[j]) * 2);
    }
    // LDS byte addresses (low 32 bits of flat shared pointer = LDS offset).
    unsigned ldsA[2][2], ldsW[2][2];
#pragma unroll
    for (int b = 0; b < 2; ++b)
#pragma unroll
        for (int j = 0; j < 2; ++j) {
            ldsA[b][j] = (unsigned)(size_t)&lA[b][rowc[j] * LDT + kcc[j]];
            ldsW[b][j] = (unsigned)(size_t)&lW[b][rowc[j] * LDT + kcc[j]];
        }

    const unsigned long long abase = (unsigned long long)A;
    const unsigned long long wbase = (unsigned long long)W;

    v8f acc[2][4] = {};

    // Fragment gather offsets per ISA 7.12.2 (wave32):
    // A: row m = lane&15; K chunks {0..7,16..23} (lanes<16) / {8..15,24..31}.
    // B: col n = lane&15; K chunks {0..7,16..23} (lanes<16) / {8..15,24..31}.
    const int kb   = (lane >> 4) * 8;
    const int koff = (lane >> 4) * 16;
    const int l15  = lane & 15;

    // issue / compute with compile-time buffer index so all LDS addresses
    // are immediates (no cndmask selection chains).
    auto issue = [&](int buf, int kstep) {
        unsigned kb2 = (unsigned)kstep * (BK * 2);
        async_ld_b128(ldsA[buf][0], aoff[0] + kb2, abase);
        async_ld_b128(ldsA[buf][1], aoff[1] + kb2, abase);
        async_ld_b128(ldsW[buf][0], woff[0] + kb2, wbase);
        async_ld_b128(ldsW[buf][1], woff[1] + kb2, wbase);
    };

    auto compute = [&](int buf) {
        const _Float16* bufA = lA[buf];
        const _Float16* bufW = lW[buf];
        v16h a[2];
#pragma unroll
        for (int mt = 0; mt < 2; ++mt) {
            int m = wm * 32 + mt * 16 + l15;
            v8h lo = *(const v8h*)&bufA[m * LDT + kb];
            v8h hi = *(const v8h*)&bufA[m * LDT + kb + 16];
            a[mt] = __builtin_shufflevector(lo, hi, 0,1,2,3,4,5,6,7,8,9,10,11,12,13,14,15);
        }
        v16h b[4];
#pragma unroll
        for (int nt = 0; nt < 4; ++nt) {
            int n = wn * 64 + nt * 16 + l15;
            v8h lo = *(const v8h*)&bufW[n * LDT + kb];
            v8h hi = *(const v8h*)&bufW[n * LDT + kb + 16];
            b[nt] = __builtin_shufflevector(lo, hi, 0,1,2,3,4,5,6,7,8,9,10,11,12,13,14,15);
        }
#pragma unroll
        for (int mt = 0; mt < 2; ++mt)
#pragma unroll
            for (int nt = 0; nt < 4; ++nt)
                acc[mt][nt] = __builtin_amdgcn_wmma_f32_16x16x32_f16(
                    false, a[mt], false, b[nt], (short)0, acc[mt][nt], false, false);
    };

    const int nk = K / BK;  // 32 or 128: even, >= 4
    issue(0, 0);
    issue(1, 1);

    // Steady state: branch-free body, 2 K-steps per iteration, 8 async
    // loads in flight (2 buffers x 4).
    for (int k = 0; k < nk - 2; k += 2) {
        wait_async_le4();          // 4 newest = buf1; buf0 complete
        __syncthreads();
        compute(0);
        __syncthreads();           // all waves done reading buf0
        issue(0, k + 2);

        wait_async_le4();          // 4 newest = buf0 refill; buf1 complete
        __syncthreads();
        compute(1);
        __syncthreads();           // all waves done reading buf1
        issue(1, k + 3);
    }

    // Epilogue: last two K-steps, nothing left to prefetch.
    wait_async_le4();
    __syncthreads();
    compute(0);
    wait_async_0();
    __syncthreads();
    compute(1);

    // C/D layout: VGPR r -> M = r + 8*(lane>=16), N = lane&15.
#pragma unroll
    for (int mt = 0; mt < 2; ++mt) {
        int mrow = m0 + wm * 32 + mt * 16 + (lane >> 4) * 8;
#pragma unroll
        for (int nt = 0; nt < 4; ++nt) {
            int col = n0 + wn * 64 + nt * 16 + l15;
#pragma unroll
            for (int r = 0; r < 8; ++r) {
                float v = acc[mt][nt][r];
                if (RELU) v = v > 0.0f ? v : 0.0f;
                C[(size_t)(mrow + r) * N + col] = (OutT)v;
            }
        }
    }
}

extern "C" void kernel_launch(void* const* d_in, const int* in_sizes, int n_in,
                              void* d_out, int out_size, void* d_ws, size_t ws_size,
                              hipStream_t stream) {
    constexpr int E = 8, T = 2048, D = 1024, H = 4096;
    const float* x   = (const float*)d_in[0];
    const float* fc1 = (const float*)d_in[1];
    const float* fc2 = (const float*)d_in[2];
    float* out = (float*)d_out;

    const size_t nx = (size_t)E * T * D;
    const size_t n1 = (size_t)E * H * D;
    const size_t n2 = (size_t)E * D * H;

    char* ws = (char*)d_ws;
    _Float16* xh   = (_Float16*)(ws);
    _Float16* w1h  = (_Float16*)(ws + nx * 2);
    _Float16* w2h  = (_Float16*)(ws + (nx + n1) * 2);
    _Float16* hbuf = (_Float16*)(ws + (nx + n1 + n2) * 2);

    cvt_f32_to_f16<<<(unsigned)(nx / 1024), 256, 0, stream>>>(x,   xh,  nx / 4);
    cvt_f32_to_f16<<<(unsigned)(n1 / 1024), 256, 0, stream>>>(fc1, w1h, n1 / 4);
    cvt_f32_to_f16<<<(unsigned)(n2 / 1024), 256, 0, stream>>>(fc2, w2h, n2 / 4);

    // h = relu(x @ fc1^T): M=T, N=H, K=D
    gemm_tn<true, _Float16><<<dim3(H / 128, T / 128, E), 256, 0, stream>>>(xh, w1h, hbuf, T, H, D);
    // out = h @ fc2^T: M=T, N=D, K=H
    gemm_tn<false, float><<<dim3(D / 128, T / 128, E), 256, 0, stream>>>(hbuf, w2h, out, T, D, H);
}
